// attention_40870908789207
// MI455X (gfx1250) — matrix operand order repaired and checked by reading, not compiled, not run on hardware
//
#include <hip/hip_runtime.h>
#include <stdint.h>

// Problem constants (B=2, C=64, H=64, W=64)
#define NPIX 8192   // B*H*W
#define CIN  64
#define DHDIM 32
#define HWSZ 4096   // H*W

typedef __attribute__((ext_vector_type(16))) __bf16 v16bf;
typedef __attribute__((ext_vector_type(8)))  float  v8f;

union V16 {
  v16bf v;
  uint4 q[2];
};

static __device__ __forceinline__ uint16_t f32_to_bf16_bits(float f) {
  uint32_t u = __float_as_uint(f);
  u += 0x7FFFu + ((u >> 16) & 1u);   // round-to-nearest-even
  return (uint16_t)(u >> 16);
}

// Async copy: 16 bytes per lane, global -> LDS, tracked by ASYNCcnt.
static __device__ __forceinline__ void async_copy_b128(uint32_t lds_off,
                                                       const void* gptr) {
  asm volatile("global_load_async_to_lds_b128 %0, %1, off"
               :
               : "v"(lds_off), "v"((unsigned long long)(uintptr_t)gptr)
               : "memory");
}
static __device__ __forceinline__ void wait_async_le1() {
  asm volatile("s_wait_asynccnt 0x1" ::: "memory");
}

static __device__ __forceinline__ uint32_t lds_off_u32(const void* p) {
  // addrspacecast(AS3->flat) is {aperture_hi, offset}; low 32 bits = LDS offset
  return (uint32_t)(uintptr_t)p;
}

// ---------------------------------------------------------------------------
// Kernel 1: 1x1-conv projections q,k,v. Emits bf16 q [N,32], k [N,32], and
// v transposed [32,N] so the PV WMMA B-operand loads are contiguous.
// ---------------------------------------------------------------------------
__global__ void qkv_proj_kernel(const float* __restrict__ x,
                                const float* __restrict__ w1, const float* __restrict__ b1,
                                const float* __restrict__ w2, const float* __restrict__ b2,
                                const float* __restrict__ w3, const float* __restrict__ b3,
                                uint16_t* __restrict__ qb, uint16_t* __restrict__ kb,
                                uint16_t* __restrict__ vbT) {
  __shared__ float sw[3][DHDIM][CIN];
  __shared__ float sb[3][DHDIM];
  int tid = threadIdx.x;
  for (int i = tid; i < 3 * DHDIM * CIN; i += 256) {
    int which = i / (DHDIM * CIN);
    int rem   = i % (DHDIM * CIN);
    const float* wsrc = (which == 0) ? w1 : ((which == 1) ? w2 : w3);
    (&sw[0][0][0])[i] = wsrc[rem];
  }
  if (tid < 3 * DHDIM) {
    int which = tid / DHDIM;
    const float* bsrc = (which == 0) ? b1 : ((which == 1) ? b2 : b3);
    sb[which][tid % DHDIM] = bsrc[tid % DHDIM];
  }
  __syncthreads();

  int n   = blockIdx.x * 256 + tid;
  int b   = n / HWSZ;
  int rem = n % HWSZ;
  const float* xp = x + (size_t)b * CIN * HWSZ + rem;
  float xc[CIN];
#pragma unroll
  for (int c = 0; c < CIN; ++c) xc[c] = xp[(size_t)c * HWSZ];

  for (int o = 0; o < DHDIM; ++o) {
    float aq = sb[0][o], ak = sb[1][o], av = sb[2][o];
#pragma unroll
    for (int c = 0; c < CIN; ++c) {
      aq = fmaf(xc[c], sw[0][o][c], aq);
      ak = fmaf(xc[c], sw[1][o][c], ak);
      av = fmaf(xc[c], sw[2][o][c], av);
    }
    qb[n * DHDIM + o]         = f32_to_bf16_bits(aq);
    kb[n * DHDIM + o]         = f32_to_bf16_bits(ak);
    vbT[(size_t)o * NPIX + n] = f32_to_bf16_bits(av);
  }
}

// ---------------------------------------------------------------------------
// Kernel 2: per-column softmax stats (softmax over axis 0 => stats per col j):
// m_j = max_i q_i.k_j, s_j = sum_i exp(q_i.k_j - m_j).
// One wave owns 16 columns (B = K^T tile is loop-invariant in regs). All 8
// waves of a block stream the SAME Q rows, so a 32-row Q chunk (2KB) is
// staged into LDS once per block with double-buffered async copies
// (waves 0-3 issue; everyone consumes after s_wait_asynccnt + barrier).
// 2 WMMAs + one online max/sum update per 32-row chunk.
// ---------------------------------------------------------------------------
__global__ void colstats_kernel(const uint16_t* __restrict__ qb,
                                const uint16_t* __restrict__ kb,
                                float* __restrict__ mcol, float* __restrict__ scol) {
  __shared__ __align__(16) uint16_t qtile[2][32 * 32];   // 2 x 2KB
  int tid  = threadIdx.x;
  int lane = tid & 31;
  int wave = tid >> 5;
  int lrow = lane & 15;
  int hi   = lane >> 4;                 // 0: K 0..7,16..23; 1: K 8..15,24..31
  int jcol = blockIdx.x * 128 + wave * 16 + lrow;

  // B operand (K^T, 32x16): lane holds kb row jcol, dh [hi*8, hi*8+8) and [16+hi*8, 24+hi*8)
  V16 bmat;
  {
    const uint4* kp = (const uint4*)(kb + (size_t)jcol * DHDIM);
    bmat.q[0] = kp[hi];
    bmat.q[1] = kp[2 + hi];
  }

  // async staging role: tids 0..127 copy one 16B segment each (32 rows x 64B)
  bool loader = tid < 128;
  int  crow   = (tid >> 2) & 31;
  int  cseg   = tid & 3;
  uint32_t ldsA = lds_off_u32(&qtile[0][crow * 32 + cseg * 8]);
  uint32_t ldsB = lds_off_u32(&qtile[1][crow * 32 + cseg * 8]);
  const uint16_t* gsrc0 = qb + (size_t)crow * DHDIM + cseg * 8;

  if (loader) async_copy_b128(ldsA, gsrc0);   // prologue: chunk 0 -> buf 0

  float m = -INFINITY, s = 0.0f;
  for (int i0 = 0; i0 < NPIX; i0 += 32) {
    int buf   = (i0 >> 5) & 1;
    int inext = (i0 + 32) & (NPIX - 1);   // wrap: keeps pipeline uniform
    if (loader)
      async_copy_b128(buf ? ldsA : ldsB, gsrc0 + (size_t)inext * DHDIM);
    wait_async_le1();                     // our older batch has landed
    __syncthreads();                      // publish LDS to all waves

    const uint16_t* qt = &qtile[buf][0];
    V16 a0, a1;                           // A tiles: rows i0..15, i0+16..31
    {
      const uint4* p0 = (const uint4*)(qt + lrow * 32);
      const uint4* p1 = (const uint4*)(qt + (16 + lrow) * 32);
      a0.q[0] = p0[hi];  a0.q[1] = p0[2 + hi];
      a1.q[0] = p1[hi];  a1.q[1] = p1[2 + hi];
    }
    v8f c0 = {}, c1 = {};
    c0 = __builtin_amdgcn_wmma_f32_16x16x32_bf16(false, a0.v, false, bmat.v,
                                                 (short)0, c0, false, false);
    c1 = __builtin_amdgcn_wmma_f32_16x16x32_bf16(false, a1.v, false, bmat.v,
                                                 (short)0, c1, false, false);
    // single online update over 32 rows (column j in lane & lane^16)
    float tmax = fmaxf(c0[0], c1[0]);
#pragma unroll
    for (int r = 1; r < 8; ++r) tmax = fmaxf(tmax, fmaxf(c0[r], c1[r]));
    tmax = fmaxf(tmax, __shfl_xor(tmax, 16, 32));
    float mnew = fmaxf(m, tmax);
    float tsum = 0.0f;
#pragma unroll
    for (int r = 0; r < 8; ++r)
      tsum += __expf(c0[r] - mnew) + __expf(c1[r] - mnew);
    tsum += __shfl_xor(tsum, 16, 32);
    s = s * __expf(m - mnew) + tsum;
    m = mnew;
    __syncthreads();                      // buffer about to be refilled
  }
  if (lane < 16) {
    mcol[jcol] = m;
    scol[jcol] = s;
  }
}

// ---------------------------------------------------------------------------
// Kernel 3: fused o = normalize(exp(QK^T)) @ V, never materializing att.
// One wave owns 16 output rows; all 8 waves stream the SAME K/V j-chunks, so
// each 32-column chunk (K: 2KB, V: 2KB) is staged into LDS once per block
// via double-buffered async copies (tid<128 -> K, tid>=128 -> V; exactly one
// b128 per thread per chunk). Per chunk: 2 S-WMMAs, exp/scale, P staged
// through wave-private LDS (C/D layout -> A layout), 2 PV-WMMAs.
// ---------------------------------------------------------------------------
__global__ void attn_out_kernel(const uint16_t* __restrict__ qb,
                                const uint16_t* __restrict__ kb,
                                const uint16_t* __restrict__ vbT,
                                const float* __restrict__ mcol,
                                const float* __restrict__ scol,
                                float* __restrict__ obuf) {
  __shared__ __align__(16) uint16_t kchunk[2][32 * 32];  // 2 x 2KB, rows j_local
  __shared__ __align__(16) uint16_t vchunk[2][32 * 32];  // 2 x 2KB, rows d
  __shared__ __align__(16) uint16_t pbuf[8][16 * 32];    // per-wave P tile (1KB)
  int tid  = threadIdx.x;
  int lane = tid & 31;
  int wave = tid >> 5;
  int lrow = lane & 15;
  int hi   = lane >> 4;
  int i0   = (blockIdx.x * 8 + wave) * 16;   // output row-tile base

  // A operand for S: this wave's 16 Q rows, loop-invariant in registers
  V16 aq;
  {
    const uint4* qp = (const uint4*)(qb + (size_t)(i0 + lrow) * DHDIM);
    aq.q[0] = qp[hi];
    aq.q[1] = qp[2 + hi];
  }

  // async staging role: tid<128 -> K chunk, tid>=128 -> V chunk
  bool isK = tid < 128;
  int  u    = tid & 127;
  int  crow = u >> 2;         // 0..31  (j_local for K, d for V)
  int  cseg = u & 3;          // 16B segment within 64B row
  uint16_t* ldsrow0 = (isK ? &kchunk[0][0] : &vchunk[0][0]) + crow * 32 + cseg * 8;
  uint16_t* ldsrow1 = (isK ? &kchunk[1][0] : &vchunk[1][0]) + crow * 32 + cseg * 8;
  uint32_t ldsA = lds_off_u32(ldsrow0);
  uint32_t ldsB = lds_off_u32(ldsrow1);
  // K row j: kb + j*32 ; V row d: vbT + d*NPIX + j0
  const uint16_t* gbase = isK ? (kb + (size_t)crow * DHDIM + cseg * 8)
                              : (vbT + (size_t)crow * NPIX + cseg * 8);
  size_t gstep = isK ? (size_t)DHDIM : (size_t)1;   // per +1 column j

  async_copy_b128(ldsA, gbase);   // prologue: chunk j0=0 -> buf 0

  uint16_t* myp = &pbuf[wave][0];
  v8f acc0 = {};   // o columns d = 0..15
  v8f acc1 = {};   // o columns d = 16..31

  for (int j0 = 0; j0 < NPIX; j0 += 32) {
    int buf   = (j0 >> 5) & 1;
    int jnext = (j0 + 32) & (NPIX - 1);
    async_copy_b128(buf ? ldsA : ldsB, gbase + (size_t)jnext * gstep);
    wait_async_le1();
    __syncthreads();

    const uint16_t* kt = &kchunk[buf][0];
    const uint16_t* vt = &vchunk[buf][0];

    // ---- S tiles for columns [j0, j0+16) and [j0+16, j0+32) ----
#pragma unroll
    for (int t = 0; t < 2; ++t) {
      int jc = j0 + t * 16 + lrow;
      V16 bk;
      {
        const uint4* kp = (const uint4*)(kt + (t * 16 + lrow) * 32);
        bk.q[0] = kp[hi];
        bk.q[1] = kp[2 + hi];
      }
      v8f sres = {};
      sres = __builtin_amdgcn_wmma_f32_16x16x32_bf16(false, aq.v, false, bk.v,
                                                     (short)0, sres, false, false);
      float mj  = mcol[jc];
      float rsj = 1.0f / scol[jc];
      int col   = t * 16 + lrow;
#pragma unroll
      for (int r = 0; r < 8; ++r) {
        float p = __expf(sres[r] - mj) * rsj;
        myp[(r + hi * 8) * 32 + col] = f32_to_bf16_bits(p);   // C/D -> row major
      }
    }
    __builtin_amdgcn_wave_barrier();   // LDS store->load order (same wave)
    // ---- read P back in A layout (16x32 bf16) ----
    V16 ap;
    {
      const uint4* pp = (const uint4*)(myp + lrow * 32);
      ap.q[0] = pp[hi];
      ap.q[1] = pp[2 + hi];
    }
    __builtin_amdgcn_wave_barrier();
    // ---- accumulate P (16x32) @ V (32x16, two d-tiles) ----
#pragma unroll
    for (int t = 0; t < 2; ++t) {
      V16 bv;
      {
        const uint4* vq = (const uint4*)(vt + (t * 16 + lrow) * 32);
        bv.q[0] = vq[hi];
        bv.q[1] = vq[2 + hi];
      }
      if (t == 0)
        acc0 = __builtin_amdgcn_wmma_f32_16x16x32_bf16(false, ap.v, false, bv.v,
                                                       (short)0, acc0, false, false);
      else
        acc1 = __builtin_amdgcn_wmma_f32_16x16x32_bf16(false, ap.v, false, bv.v,
                                                       (short)0, acc1, false, false);
    }
    __syncthreads();                   // buffers about to be refilled
  }
  // writeback: C/D layout -> obuf [N, 32] fp32
#pragma unroll
  for (int r = 0; r < 8; ++r) {
    int row = i0 + r + hi * 8;
    obuf[(size_t)row * DHDIM + lrow]      = acc0[r];
    obuf[(size_t)row * DHDIM + 16 + lrow] = acc1[r];
  }
}

// ---------------------------------------------------------------------------
// Kernel 4: y = o @ w4^T + b4 + x  (1x1 conv back to C=64 + residual)
// ---------------------------------------------------------------------------
__global__ void out_proj_kernel(const float* __restrict__ obuf,
                                const float* __restrict__ w4, const float* __restrict__ b4,
                                const float* __restrict__ x, float* __restrict__ y) {
  int t  = blockIdx.x * 256 + threadIdx.x;
  int n  = t & (NPIX - 1);
  int oc = t >> 13;                    // NPIX == 2^13
  const float* orow = obuf + (size_t)n * DHDIM;
  const float* wrow = w4 + (size_t)oc * DHDIM;
  float acc = b4[oc];
#pragma unroll
  for (int d = 0; d < DHDIM; ++d) acc = fmaf(orow[d], wrow[d], acc);
  int b = n / HWSZ, rem = n % HWSZ;
  size_t xi = (size_t)b * CIN * HWSZ + (size_t)oc * HWSZ + rem;
  y[xi] = acc + x[xi];
}

// ---------------------------------------------------------------------------
extern "C" void kernel_launch(void* const* d_in, const int* in_sizes, int n_in,
                              void* d_out, int out_size, void* d_ws, size_t ws_size,
                              hipStream_t stream) {
  const float* x  = (const float*)d_in[0];
  const float* w1 = (const float*)d_in[1];
  const float* b1 = (const float*)d_in[2];
  const float* w2 = (const float*)d_in[3];
  const float* b2 = (const float*)d_in[4];
  const float* w3 = (const float*)d_in[5];
  const float* b3 = (const float*)d_in[6];
  const float* w4 = (const float*)d_in[7];
  const float* b4 = (const float*)d_in[8];
  float* y = (float*)d_out;

  uint8_t* ws = (uint8_t*)d_ws;
  uint16_t* qb   = (uint16_t*)(ws);                       // 512 KB  (8192x32 bf16)
  uint16_t* kb   = (uint16_t*)(ws + 512u * 1024u);        // 512 KB
  uint16_t* vbT  = (uint16_t*)(ws + 1024u * 1024u);       // 512 KB  (32x8192 bf16)
  float*    mcol = (float*)(ws + 1536u * 1024u);          // 32 KB
  float*    scol = (float*)(ws + 1568u * 1024u);          // 32 KB
  float*    obuf = (float*)(ws + 1600u * 1024u);          // 1 MB    (8192x32 f32)

  qkv_proj_kernel<<<NPIX / 256, 256, 0, stream>>>(x, w1, b1, w2, b2, w3, b3,
                                                  qb, kb, vbT);
  colstats_kernel<<<NPIX / 128, 256, 0, stream>>>(qb, kb, mcol, scol);
  attn_out_kernel<<<NPIX / 128, 256, 0, stream>>>(qb, kb, vbT, mcol, scol, obuf);
  out_proj_kernel<<<(NPIX * CIN) / 256, 256, 0, stream>>>(obuf, w4, b4, x, y);
}
